// GAT_28475633173050
// MI455X (gfx1250) — matrix operand order repaired and checked by reading, not compiled, not run on hardware
//
#include <hip/hip_runtime.h>
#include <hip/hip_bf16.h>

// ---------------------------------------------------------------------------
// 2-layer GAT for gfx1250 (MI455X). Dense projections use v_wmma_f32_16x16x32_f16.
// ---------------------------------------------------------------------------

typedef __attribute__((ext_vector_type(16))) _Float16 v16h;
typedef __attribute__((ext_vector_type(8)))  _Float16 v8h;
typedef __attribute__((ext_vector_type(8)))  float    v8f;

#define DEVFN __device__ __forceinline__

DEVFN void atomicMaxF(float* addr, float val) {
  // sign-split trick: works for mixed-sign values when initialized to -inf
  if (val >= 0.f) atomicMax((int*)addr, __float_as_int(val));
  else            atomicMin((unsigned int*)addr, __float_as_uint(val));
}

// ----------------------------- utility kernels -----------------------------

__global__ void gat_fill(float* __restrict__ p, float v, long n) {
  long i = (long)blockIdx.x * blockDim.x + threadIdx.x;
  if (i < n) p[i] = v;
}

__global__ void gat_cvt_f16(const float* __restrict__ in, _Float16* __restrict__ out, long n) {
  long i = (long)blockIdx.x * blockDim.x + threadIdx.x;
  if (i < n) out[i] = (_Float16)in[i];
}

// Wt[nout][k] = (f16) W[k][nout]   (column-major f16 copy of weight)
__global__ void gat_tcvt(const float* __restrict__ W, _Float16* __restrict__ Wt,
                         int K, int Nout) {
  long t = (long)blockIdx.x * blockDim.x + threadIdx.x;
  if (t >= (long)K * Nout) return;
  int k = (int)(t / Nout), n = (int)(t % Nout);
  Wt[(size_t)n * K + k] = (_Float16)W[t];
}

// block-reduced sum of edge_attr rows (ED=6)
__global__ void gat_ea_sum(const float* __restrict__ ea, float* __restrict__ acc, int E) {
  __shared__ float s[6];
  if (threadIdx.x < 6) s[threadIdx.x] = 0.f;
  __syncthreads();
  float loc[6] = {0.f, 0.f, 0.f, 0.f, 0.f, 0.f};
  for (long i = (long)blockIdx.x * blockDim.x + threadIdx.x; i < E;
       i += (long)gridDim.x * blockDim.x) {
#pragma unroll
    for (int d = 0; d < 6; ++d) loc[d] += ea[i * 6 + d];
  }
#pragma unroll
  for (int d = 0; d < 6; ++d) atomicAdd(&s[d], loc[d]);
  __syncthreads();
  if (threadIdx.x < 6) atomicAdd(&acc[threadIdx.x], s[threadIdx.x]);
}

// tiny prep: ea_mean, q1[6][4], q2[6], self-loop edge scores
__global__ void gat_prep(const float* __restrict__ ea_acc, float invE,
                         const float* __restrict__ We1, const float* __restrict__ ae1,
                         const float* __restrict__ We2, const float* __restrict__ ae2,
                         float* __restrict__ ea_mean, float* __restrict__ q1,
                         float* __restrict__ q2, float* __restrict__ se1,
                         float* __restrict__ se2) {
  int t = threadIdx.x;
  if (t < 6) ea_mean[t] = ea_acc[t] * invE;
  __syncthreads();
  if (t < 24) {                       // q1[d*4+h]
    int d = t >> 2, h = t & 3;
    float s = 0.f;
    for (int c = 0; c < 64; ++c) s += We1[d * 256 + h * 64 + c] * ae1[h * 64 + c];
    q1[t] = s;
  }
  if (t >= 32 && t < 38) {            // q2[d]
    int d = t - 32;
    float s = 0.f;
    for (int c = 0; c < 64; ++c) s += We2[d * 64 + c] * ae2[c];
    q2[d] = s;
  }
  __syncthreads();
  if (t < 4) { float s = 0.f; for (int d = 0; d < 6; ++d) s += ea_mean[d] * q1[d * 4 + t]; se1[t] = s; }
  if (t == 8) { float s = 0.f; for (int d = 0; d < 6; ++d) s += ea_mean[d] * q2[d]; se2[0] = s; }
}

// ------------------------------ WMMA GEMM ----------------------------------
// C[M x Nout] = A[M x K] * B[K x Nout];  A is f16 row-major, Bt is f16 [Nout x K]
// (B column-major).  One 16x16 output tile per wave32; blockDim = (32, Nout/16).
// VGPR packings: A and B both use contiguous K per lane half (k = 16*(lane/16) + s).
__global__ void gat_wmma_gemm(const _Float16* __restrict__ A,
                              const _Float16* __restrict__ Bt,
                              float* __restrict__ C, int K, int Nout) {
  const int lane  = threadIdx.x;        // 0..31
  const int tileN = threadIdx.y;        // column tile
  const int tileM = blockIdx.x;         // row tile
  const int r15   = lane & 15;
  const int row   = tileM * 16 + r15;
  const int col   = tileN * 16 + r15;
  const int kbA   = (lane < 16) ? 0 : 16;   // A: 16 contiguous K per lane
  const int kbB   = (lane < 16) ? 0 : 16;   // B: 16 contiguous K per lane

  const _Float16* ap = A  + (size_t)row * K + kbA;
  const _Float16* bp = Bt + (size_t)col * K + kbB;

  v8f acc = {};
#pragma unroll 4
  for (int k = 0; k < K; k += 32) {
    __builtin_prefetch(ap + k + 128, 0, 0);   // global_prefetch_b8
    v8h a0 = *(const v8h*)(ap + k);
    v8h a1 = *(const v8h*)(ap + k + 8);
    v16h av = __builtin_shufflevector(a0, a1, 0, 1, 2, 3, 4, 5, 6, 7,
                                      8, 9, 10, 11, 12, 13, 14, 15);
    v16h bv = *(const v16h*)(bp + k);
    acc = __builtin_amdgcn_wmma_f32_16x16x32_f16(
        /*neg_a=*/false, av, /*neg_b=*/false, bv,
        /*c_mod=*/(short)0, acc, /*reuse_a=*/false, /*reuse_b=*/false);
  }
  // C/D layout: lane l: N = l&15; VGPR v: M = v + 8*(l>=16)
  float* cp = C + (size_t)(tileM * 16 + (lane >> 4) * 8) * Nout + tileN * 16 + r15;
#pragma unroll
  for (int v = 0; v < 8; ++v) cp[(size_t)v * Nout] = acc[v];
}

// ------------------------------ attention ----------------------------------

// per-node src/dst scores: als[n,h] = <h[n,h,:], a_src[h,:]>
__global__ void gat_scores(const float* __restrict__ h,
                           const float* __restrict__ a_src,
                           const float* __restrict__ a_dst,
                           float* __restrict__ als, float* __restrict__ ald,
                           int N, int H, int Cc) {
  long t = (long)blockIdx.x * blockDim.x + threadIdx.x;
  if (t >= (long)N * H) return;
  int n = (int)(t / H), hh = (int)(t % H);
  const float4* hp = (const float4*)(h + (size_t)n * H * Cc + (size_t)hh * Cc);
  const float4* as = (const float4*)(a_src + (size_t)hh * Cc);
  const float4* ad = (const float4*)(a_dst + (size_t)hh * Cc);
  float s = 0.f, d = 0.f;
  for (int i = 0; i < Cc / 4; ++i) {
    float4 v = hp[i], a = as[i], b = ad[i];
    s += v.x * a.x + v.y * a.y + v.z * a.z + v.w * a.w;
    d += v.x * b.x + v.y * b.y + v.z * b.z + v.w * b.w;
  }
  als[t] = s; ald[t] = d;
}

// per-edge logits (leaky-relu 0.2) + atomic segment max over dst
__global__ void gat_logits_max(const int* __restrict__ src, const int* __restrict__ dst,
                               const float* __restrict__ ea, const float* __restrict__ q,
                               const float* __restrict__ se,
                               const float* __restrict__ als, const float* __restrict__ ald,
                               float* __restrict__ logits, float* __restrict__ mx,
                               int E, int N, int H) {
  long e = (long)blockIdx.x * blockDim.x + threadIdx.x;
  if (e >= (long)E + N) return;
  int s, d;
  float ale[4];
  if (e < E) {
    s = src[e]; d = dst[e];
    for (int h = 0; h < H; ++h) ale[h] = 0.f;
    for (int k = 0; k < 6; ++k) {
      float v = ea[e * 6 + k];
      for (int h = 0; h < H; ++h) ale[h] += v * q[k * H + h];
    }
  } else {
    s = d = (int)(e - E);
    for (int h = 0; h < H; ++h) ale[h] = se[h];
  }
  for (int h = 0; h < H; ++h) {
    float l = als[(size_t)s * H + h] + ald[(size_t)d * H + h] + ale[h];
    l = (l > 0.f) ? l : 0.2f * l;            // leaky_relu(0.2)
    logits[e * H + h] = l;
    atomicMaxF(&mx[(size_t)d * H + h], l);
  }
}

// p = exp(logit - max[dst]) stored in place; denom[dst] += p
__global__ void gat_softmax_num(const int* __restrict__ dst, const float* __restrict__ mx,
                                float* __restrict__ logits, float* __restrict__ den,
                                int E, int N, int H) {
  long t = (long)blockIdx.x * blockDim.x + threadIdx.x;
  if (t >= ((long)E + N) * H) return;
  long e = t / H; int h = (int)(t % H);
  int d = (e < E) ? dst[e] : (int)(e - E);
  float p = __expf(logits[t] - mx[(size_t)d * H + h]);
  logits[t] = p;
  atomicAdd(&den[(size_t)d * H + h], p);
}

// out[dst] += h[src] * alpha  (float4 gather + 4 atomic f32 adds per thread)
__global__ void gat_scatter(const int* __restrict__ src, const int* __restrict__ dst,
                            const float* __restrict__ hsrc, const float* __restrict__ p,
                            const float* __restrict__ den, float* __restrict__ out,
                            int E, int N, int H, int Cc) {
  const int F = H * Cc;
  const int tpe = F / 4;
  long t = (long)blockIdx.x * blockDim.x + threadIdx.x;
  long e = t / tpe;
  if (e >= (long)E + N) return;
  int c0 = (int)(t % tpe) * 4;
  int s = (e < E) ? src[e] : (int)(e - E);
  int d = (e < E) ? dst[e] : (int)(e - E);
  int h = c0 / Cc;
  float alpha = p[e * H + h] / (den[(size_t)d * H + h] + 1e-16f);
  float4 v = *(const float4*)(hsrc + (size_t)s * F + c0);
  float* o = out + (size_t)d * F + c0;
  atomicAdd(o + 0, v.x * alpha);
  atomicAdd(o + 1, v.y * alpha);
  atomicAdd(o + 2, v.z * alpha);
  atomicAdd(o + 3, v.w * alpha);
}

// y = LN(relu(y + b)) * w + bb ; one node per wave32, wave shuffle reductions
__global__ void gat_bias_relu_ln(float* __restrict__ y, const float* __restrict__ b,
                                 const float* __restrict__ w, const float* __restrict__ bb,
                                 int N, int F) {
  int n = blockIdx.x * blockDim.y + threadIdx.y;
  if (n >= N) return;
  int lane = threadIdx.x;
  int per = F / 32;                       // 8 (F=256) or 2 (F=64)
  float vals[8];
  float s = 0.f;
  for (int i = 0; i < per; ++i) {
    int c = lane * per + i;
    float v = y[(size_t)n * F + c] + b[c];
    v = fmaxf(v, 0.f);
    vals[i] = v; s += v;
  }
#pragma unroll
  for (int off = 16; off >= 1; off >>= 1) s += __shfl_xor(s, off, 32);
  float mu = s / (float)F;
  float vs = 0.f;
  for (int i = 0; i < per; ++i) { float d = vals[i] - mu; vs += d * d; }
#pragma unroll
  for (int off = 16; off >= 1; off >>= 1) vs += __shfl_xor(vs, off, 32);
  float inv = rsqrtf(vs / (float)F + 1e-5f);
  for (int i = 0; i < per; ++i) {
    int c = lane * per + i;
    y[(size_t)n * F + c] = (vals[i] - mu) * inv * w[c] + bb[c];
  }
}

// ------------------------------ pooling ------------------------------------

__global__ void gat_pool_sum(const float* __restrict__ h, const int* __restrict__ batch,
                             float* __restrict__ sums, float* __restrict__ cnt,
                             int N, int F) {
  long t = (long)blockIdx.x * blockDim.x + threadIdx.x;
  if (t >= (long)N * (F / 4)) return;
  int n = (int)(t / (F / 4)), j = (int)(t % (F / 4));
  int g = batch[n];
  float4 v = *(const float4*)(h + (size_t)n * F + j * 4);
  float* o = sums + (size_t)g * F + j * 4;
  atomicAdd(o + 0, v.x); atomicAdd(o + 1, v.y);
  atomicAdd(o + 2, v.z); atomicAdd(o + 3, v.w);
  if (j == 0) atomicAdd(&cnt[g], 1.f);
}

__global__ void gat_pool_div(const float* __restrict__ sums, const float* __restrict__ cnt,
                             float* __restrict__ out, int G, int F) {
  int t = blockIdx.x * blockDim.x + threadIdx.x;
  if (t >= G * F) return;
  out[t] = sums[t] / fmaxf(cnt[t / F], 1.f);
}

// ------------------------------ launcher -----------------------------------

static inline unsigned cdiv(long n, int b) { return (unsigned)((n + b - 1) / b); }

extern "C" void kernel_launch(void* const* d_in, const int* in_sizes, int n_in,
                              void* d_out, int out_size, void* d_ws, size_t ws_size,
                              hipStream_t stream) {
  const int F_IN = 128, H1 = 4, C1 = 64, F1 = 256, C2 = 64, G = 64;
  const int Nn = in_sizes[0] / F_IN;
  const int Ee = in_sizes[1] / 2;

  const float* x     = (const float*)d_in[0];
  const int*   ei    = (const int*)d_in[1];
  const int*   src   = ei;
  const int*   dst   = ei + Ee;
  const float* eattr = (const float*)d_in[2];
  const int*   batch = (const int*)d_in[3];
  const float* W1    = (const float*)d_in[4];
  const float* We1   = (const float*)d_in[5];
  const float* asrc1 = (const float*)d_in[6];
  const float* adst1 = (const float*)d_in[7];
  const float* aedg1 = (const float*)d_in[8];
  const float* b1    = (const float*)d_in[9];
  const float* ln1w  = (const float*)d_in[10];
  const float* ln1b  = (const float*)d_in[11];
  const float* W2    = (const float*)d_in[12];
  const float* We2   = (const float*)d_in[13];
  const float* asrc2 = (const float*)d_in[14];
  const float* adst2 = (const float*)d_in[15];
  const float* aedg2 = (const float*)d_in[16];
  const float* b2    = (const float*)d_in[17];
  const float* ln2w  = (const float*)d_in[18];
  const float* ln2b  = (const float*)d_in[19];
  (void)n_in; (void)out_size; (void)ws_size;

  // ---- workspace layout ----
  char* wsb = (char*)d_ws;
  size_t off = 0;
  auto alloc = [&](size_t bytes) -> void* {
    off = (off + 255) & ~(size_t)255;
    void* p = wsb + off; off += bytes; return p;
  };
  float* ea_acc  = (float*)alloc(6 * 4);
  float* ea_mean = (float*)alloc(6 * 4);
  float* q1      = (float*)alloc(24 * 4);
  float* q2      = (float*)alloc(6 * 4);
  float* se1     = (float*)alloc(4 * 4);
  float* se2     = (float*)alloc(4);
  _Float16* xh   = (_Float16*)alloc((size_t)Nn * F_IN * 2);
  _Float16* w1t  = (_Float16*)alloc((size_t)F1 * F_IN * 2);
  _Float16* w2t  = (_Float16*)alloc((size_t)C2 * F1 * 2);
  float* h1      = (float*)alloc((size_t)Nn * F1 * 4);
  float* als1    = (float*)alloc((size_t)Nn * H1 * 4);
  float* ald1    = (float*)alloc((size_t)Nn * H1 * 4);
  float* m1      = (float*)alloc((size_t)Nn * H1 * 4);
  float* den1    = (float*)alloc((size_t)Nn * H1 * 4);
  float* logit1  = (float*)alloc((size_t)(Ee + Nn) * H1 * 4);
  float* out1    = (float*)alloc((size_t)Nn * F1 * 4);
  _Float16* h1h  = (_Float16*)alloc((size_t)Nn * F1 * 2);
  float* h2      = (float*)alloc((size_t)Nn * C2 * 4);
  float* als2    = (float*)alloc((size_t)Nn * 4);
  float* ald2    = (float*)alloc((size_t)Nn * 4);
  float* m2      = (float*)alloc((size_t)Nn * 4);
  float* den2    = (float*)alloc((size_t)Nn * 4);
  float* logit2  = (float*)alloc((size_t)(Ee + Nn) * 4);
  float* out2    = (float*)alloc((size_t)Nn * C2 * 4);
  float* sums    = (float*)alloc((size_t)G * C2 * 4);
  float* cnt     = (float*)alloc((size_t)G * 4);

  const float NEG_INF = -__builtin_inff();
  const int TB = 256;

  // ---- init accumulators (every call: replay-safe) ----
  gat_fill<<<1, 64, 0, stream>>>(ea_acc, 0.f, 6);
  gat_fill<<<cdiv((long)Nn * H1, TB), TB, 0, stream>>>(m1, NEG_INF, (long)Nn * H1);
  gat_fill<<<cdiv((long)Nn * H1, TB), TB, 0, stream>>>(den1, 0.f, (long)Nn * H1);
  gat_fill<<<cdiv((long)Nn * F1, TB), TB, 0, stream>>>(out1, 0.f, (long)Nn * F1);
  gat_fill<<<cdiv((long)Nn, TB), TB, 0, stream>>>(m2, NEG_INF, (long)Nn);
  gat_fill<<<cdiv((long)Nn, TB), TB, 0, stream>>>(den2, 0.f, (long)Nn);
  gat_fill<<<cdiv((long)Nn * C2, TB), TB, 0, stream>>>(out2, 0.f, (long)Nn * C2);
  gat_fill<<<cdiv((long)G * C2, TB), TB, 0, stream>>>(sums, 0.f, (long)G * C2);
  gat_fill<<<1, 64, 0, stream>>>(cnt, 0.f, G);

  // ---- small precompute ----
  gat_ea_sum<<<256, TB, 0, stream>>>(eattr, ea_acc, Ee);
  gat_prep<<<1, 64, 0, stream>>>(ea_acc, 1.f / (float)Ee, We1, aedg1, We2, aedg2,
                                 ea_mean, q1, q2, se1, se2);

  // ---- f16 conversions / weight transposes ----
  gat_cvt_f16<<<cdiv((long)Nn * F_IN, TB), TB, 0, stream>>>(x, xh, (long)Nn * F_IN);
  gat_tcvt<<<cdiv((long)F_IN * F1, TB), TB, 0, stream>>>(W1, w1t, F_IN, F1);
  gat_tcvt<<<cdiv((long)F1 * C2, TB), TB, 0, stream>>>(W2, w2t, F1, C2);

  // ---- layer 1 ----
  gat_wmma_gemm<<<dim3(Nn / 16), dim3(32, F1 / 16), 0, stream>>>(xh, w1t, h1, F_IN, F1);
  gat_scores<<<cdiv((long)Nn * H1, TB), TB, 0, stream>>>(h1, asrc1, adst1, als1, ald1, Nn, H1, C1);
  gat_logits_max<<<cdiv((long)Ee + Nn, TB), TB, 0, stream>>>(src, dst, eattr, q1, se1,
                                                             als1, ald1, logit1, m1, Ee, Nn, H1);
  gat_softmax_num<<<cdiv(((long)Ee + Nn) * H1, TB), TB, 0, stream>>>(dst, m1, logit1, den1, Ee, Nn, H1);
  gat_scatter<<<cdiv(((long)Ee + Nn) * (F1 / 4), TB), TB, 0, stream>>>(src, dst, h1, logit1, den1,
                                                                       out1, Ee, Nn, H1, C1);
  gat_bias_relu_ln<<<cdiv((long)Nn, 4), dim3(32, 4), 0, stream>>>(out1, b1, ln1w, ln1b, Nn, F1);

  // ---- layer 2 ----
  gat_cvt_f16<<<cdiv((long)Nn * F1, TB), TB, 0, stream>>>(out1, h1h, (long)Nn * F1);
  gat_wmma_gemm<<<dim3(Nn / 16), dim3(32, C2 / 16), 0, stream>>>(h1h, w2t, h2, F1, C2);
  gat_scores<<<cdiv((long)Nn, TB), TB, 0, stream>>>(h2, asrc2, adst2, als2, ald2, Nn, 1, C2);
  gat_logits_max<<<cdiv((long)Ee + Nn, TB), TB, 0, stream>>>(src, dst, eattr, q2, se2,
                                                             als2, ald2, logit2, m2, Ee, Nn, 1);
  gat_softmax_num<<<cdiv((long)Ee + Nn, TB), TB, 0, stream>>>(dst, m2, logit2, den2, Ee, Nn, 1);
  gat_scatter<<<cdiv(((long)Ee + Nn) * (C2 / 4), TB), TB, 0, stream>>>(src, dst, h2, logit2, den2,
                                                                       out2, Ee, Nn, 1, C2);
  gat_bias_relu_ln<<<cdiv((long)Nn, 4), dim3(32, 4), 0, stream>>>(out2, b2, ln2w, ln2b, Nn, C2);

  // ---- mean pool per graph ----
  gat_pool_sum<<<cdiv((long)Nn * (C2 / 4), TB), TB, 0, stream>>>(out2, batch, sums, cnt, Nn, C2);
  gat_pool_div<<<cdiv((long)G * C2, TB), TB, 0, stream>>>(sums, cnt, (float*)d_out, G, C2);
}
